// ReformerDec_4698694222593
// MI455X (gfx1250) — compile-verified, matrix operand order checked by reading, not run on hardware
//
#include <hip/hip_runtime.h>
#include <hip/hip_bf16.h>

typedef __attribute__((ext_vector_type(16))) _Float16 v16h;
typedef __attribute__((ext_vector_type(8)))  _Float16 v8h;
typedef __attribute__((ext_vector_type(4)))  _Float16 v4h;
typedef __attribute__((ext_vector_type(8)))  float    v8f;

#define BQ 2
#define TSELF 4096
#define TKV 5120
#define DIMC 512
#define NHEADS 8
#define BH 16          // BQ * NHEADS
#define DH 64

__device__ inline v8f wmma16(v16h a, v16h b, v8f c) {
  return __builtin_amdgcn_wmma_f32_16x16x32_f16(false, a, false, b, (short)0, c,
                                                false, false);
}

// 16-bit A/B fragment from a row-major [m][k] LDS tile.
// ISA layout: lane m=lane&15, hi=lane>>4; the 16 halves a lane needs are the
// two contiguous runs K=[k0+hi*8, +8) and K=[k0+16+hi*8, +8)  -> 2x ds_load_b128.
__device__ inline v16h load_frag(const _Float16* base, int stride, int k0) {
  int lane = threadIdx.x & 31;
  int m = lane & 15, hi = lane >> 4;
  const _Float16* p = base + m * stride + k0 + hi * 8;
  v8h r0 = *(const v8h*)p;
  v8h r1 = *(const v8h*)(p + 16);
  return __builtin_shufflevector(r0, r1, 0, 1, 2, 3, 4, 5, 6, 7,
                                 8, 9, 10, 11, 12, 13, 14, 15);
}

// branchless fast tanh via v_exp_f32 (overflow-safe)
__device__ inline float fast_tanh(float u) {
  float t = __expf(-2.f * fabsf(u));
  float r = (1.f - t) / (1.f + t);
  return copysignf(r, u);
}

// ---------------------------------------------------------------- elementwise
__global__ void copy_kernel(float* __restrict__ dst, const float* __restrict__ src, long n) {
  long i = (long)blockIdx.x * blockDim.x + threadIdx.x;
  if (i < n) dst[i] = src[i];
}

__global__ void final_add_kernel(float* __restrict__ out, const float* __restrict__ a,
                                 const float* __restrict__ b, long n) {
  long i = (long)blockIdx.x * blockDim.x + threadIdx.x;
  if (i < n) out[i] = a[i] + b[i];
}

// x1[b, tt, c] += proj[b*T + tt, c]  (truncate T rows -> t rows)
__global__ void add_trunc_kernel(float* __restrict__ x1, const float* __restrict__ proj, int T) {
  long i = (long)blockIdx.x * blockDim.x + threadIdx.x;
  long n = (long)BQ * TSELF * DIMC;
  if (i >= n) return;
  int b = (int)(i / ((long)TSELF * DIMC));
  long rem = i % ((long)TSELF * DIMC);
  int tt = (int)(rem / DIMC), c = (int)(rem % DIMC);
  x1[i] += proj[((long)b * T + tt) * DIMC + c];
}

__global__ void copy_keys_kernel(float* __restrict__ xx, const float* __restrict__ keys) {
  long i = (long)blockIdx.x * blockDim.x + threadIdx.x;
  long n = (long)BQ * 1024 * DIMC;
  if (i >= n) return;
  int b = (int)(i / (1024 * DIMC));
  long rem = i % (1024 * DIMC);
  int r = (int)(rem / DIMC), c = (int)(rem % DIMC);
  xx[((long)b * TKV + TSELF + r) * DIMC + c] = keys[i];
}

// ----------------------------------------------------------------- layernorm
__global__ __launch_bounds__(128)
void ln_kernel(const float* __restrict__ x, const float* __restrict__ g,
               const float* __restrict__ b, float* __restrict__ out,
               int rows_in_per_batch, int rows_out_per_batch) {
  int r = blockIdx.x;
  int bat = r / rows_in_per_batch, tt = r % rows_in_per_batch;
  const float* xr = x + (long)r * DIMC;
  float* orow = out + ((long)bat * rows_out_per_batch + tt) * DIMC;
  __shared__ float s1[128], s2[128];
  float ls = 0.f, lq = 0.f;
  for (int i = threadIdx.x; i < DIMC; i += 128) {
    float v = xr[i];
    ls += v; lq += v * v;
  }
  s1[threadIdx.x] = ls; s2[threadIdx.x] = lq;
  __syncthreads();
  for (int s = 64; s > 0; s >>= 1) {
    if (threadIdx.x < s) { s1[threadIdx.x] += s1[threadIdx.x + s]; s2[threadIdx.x] += s2[threadIdx.x + s]; }
    __syncthreads();
  }
  float mean = s1[0] * (1.f / DIMC);
  float var = s2[0] * (1.f / DIMC) - mean * mean;
  float inv = rsqrtf(var + 1e-5f);
  for (int i = threadIdx.x; i < DIMC; i += 128)
    orow[i] = (xr[i] - mean) * inv * g[i] + b[i];
}

// ------------------------------------------------------------------ WMMA GEMM
// C[M,N] = act(A[M,K] @ B[K,N] + bias) + residual   (fp32 IO, fp16 WMMA)
// M, N multiples of 64; K multiple of 32.
__global__ __launch_bounds__(256)
void gemm_wmma_kernel(const float* __restrict__ A, const float* __restrict__ B,
                      const float* __restrict__ bias, float* __restrict__ C,
                      const float* __restrict__ residual,
                      int M, int N, int K, int act) {
  __shared__ _Float16 Ah[64 * 40];   // [m][k]  stride 40 halves (80B, 16B-aligned)
  __shared__ _Float16 Bt[64 * 40];   // [n][k]  transposed B tile
  int m0 = blockIdx.y * 64, n0 = blockIdx.x * 64;
  int tid = threadIdx.x;
  int wave = tid >> 5, lane = tid & 31;
  int mi = wave & 3, nbase = (wave >> 2) << 1;
  v8f acc0 = {}, acc1 = {};

  for (int k0 = 0; k0 < K; k0 += 32) {
    {   // A tile: 64x32, vectorized float4 -> v8h b128 store
      int r = tid >> 2, c0 = (tid & 3) * 8;
      const float* src = A + (long)(m0 + r) * K + k0 + c0;
      if (k0 + 64 < K) __builtin_prefetch(src + 32, 0, 1);
      float4 f0 = *(const float4*)src;
      float4 f1 = *(const float4*)(src + 4);
      v8h h;
      h[0] = (_Float16)f0.x; h[1] = (_Float16)f0.y; h[2] = (_Float16)f0.z; h[3] = (_Float16)f0.w;
      h[4] = (_Float16)f1.x; h[5] = (_Float16)f1.y; h[6] = (_Float16)f1.z; h[7] = (_Float16)f1.w;
      *(v8h*)(Ah + r * 40 + c0) = h;
    }
    {   // B tile: 32x64, transposed into Bt[n][k]
      int kr = tid >> 3, c0 = (tid & 7) * 8;
      const float* src = B + (long)(k0 + kr) * N + n0 + c0;
      float4 f0 = *(const float4*)src;
      float4 f1 = *(const float4*)(src + 4);
      float f[8] = {f0.x, f0.y, f0.z, f0.w, f1.x, f1.y, f1.z, f1.w};
#pragma unroll
      for (int j = 0; j < 8; ++j)
        Bt[(c0 + j) * 40 + kr] = (_Float16)f[j];
    }
    __syncthreads();
    v16h a  = load_frag(Ah + mi * 16 * 40, 40, 0);
    v16h b0 = load_frag(Bt + nbase * 16 * 40, 40, 0);
    acc0 = wmma16(a, b0, acc0);
    v16h b1 = load_frag(Bt + (nbase + 1) * 16 * 40, 40, 0);
    acc1 = wmma16(a, b1, acc1);
    __syncthreads();
  }

  int nn = lane & 15, hi = lane >> 4;
#pragma unroll
  for (int tile = 0; tile < 2; ++tile) {
    v8f acc = tile ? acc1 : acc0;
    int gn = n0 + (nbase + tile) * 16 + nn;
    float bv = bias ? bias[gn] : 0.f;
#pragma unroll
    for (int v = 0; v < 8; ++v) {
      int gm = m0 + mi * 16 + v + 8 * hi;
      float val = acc[v] + bv;
      if (act == 1) {  // jax.nn.gelu (tanh approx, jax default)
        float u = 0.7978845608028654f * (val + 0.044715f * val * val * val);
        val = 0.5f * val * (1.f + fast_tanh(u));
      }
      if (residual) val += residual[(long)gm * N + gn];
      C[(long)gm * N + gn] = val;
    }
  }
}

// --------------------------------------------------------------- rotation RNG
__device__ inline unsigned long long splitmix(unsigned long long z) {
  z += 0x9E3779B97F4A7C15ull;
  z = (z ^ (z >> 30)) * 0xBF58476D1CE4E5B9ull;
  z = (z ^ (z >> 27)) * 0x94D049BB133111EBull;
  return z ^ (z >> 31);
}

__global__ void rotgen_kernel(float* __restrict__ rot, int n, int seed) {
  int i = blockIdx.x * blockDim.x + threadIdx.x;
  if (i >= n) return;
  unsigned long long z = ((unsigned long long)(seed + 7) << 32) ^ (unsigned long long)i;
  unsigned long long h1 = splitmix(z), h2 = splitmix(z ^ 0xDEADBEEFCAFEF00Dull);
  float u1 = (float)((h1 >> 11) + 1ull) * (1.0f / 9007199254740992.0f);
  float u2 = (float)(h2 >> 11) * (1.0f / 9007199254740992.0f);
  rot[i] = sqrtf(-2.f * __logf(u1)) * __cosf(6.283185307f * u2);
}

// ------------------------------------------------------------- LSH bucketing
__global__ void bucket_kernel(const float* __restrict__ qkb, const float* __restrict__ rot,
                              int* __restrict__ buckets, int T, int nr, int nb) {
  int idx = blockIdx.x * blockDim.x + threadIdx.x;
  if (idx >= BH * 2 * T) return;
  int t = idx % T;
  int h = (idx / T) & 1;
  int row = idx / (2 * T);
  int b = row >> 3, hd = row & 7;
  const float* q = qkb + ((long)b * T + t) * DIMC + hd * DH;
  float qr[DH];
#pragma unroll
  for (int d = 0; d < DH; ++d) qr[d] = q[d];
  float best = -3.0e38f; int bi = 0;
  for (int j = 0; j < nr; ++j) {
    float dot = 0.f;
#pragma unroll
    for (int d = 0; d < DH; ++d) dot += qr[d] * rot[(d * 2 + h) * nr + j];
    if (dot > best)  { best = dot;  bi = j; }
    if (-dot > best) { best = -dot; bi = nr + j; }
  }
  buckets[(long)row * (2 * T) + h * T + t] = bi + h * nb;
}

// ------------------------------------------------------- bitonic argsort/row
__global__ __launch_bounds__(1024)
void sort_kernel(const int* __restrict__ buckets, int* __restrict__ sticker,
                 int* __restrict__ undo, int Nreal, int npad, int T) {
  extern __shared__ char smem[];
  unsigned long long* key = (unsigned long long*)smem;
  int row = blockIdx.x;
  const int* br = buckets + (long)row * Nreal;
  for (int i = threadIdx.x; i < npad; i += 1024) {
    unsigned long long k;
    if (i < Nreal) {
      unsigned long long bk = (unsigned long long)(unsigned)br[i];
      k = ((bk * (unsigned)T + (unsigned)(i % T)) << 14) | (unsigned)i;
    } else {
      k = ~0ull;
    }
    key[i] = k;
  }
  __syncthreads();
  for (int size = 2; size <= npad; size <<= 1) {
    for (int stride = size >> 1; stride > 0; stride >>= 1) {
      for (int i = threadIdx.x; i < (npad >> 1); i += 1024) {
        int pos = i & (stride - 1);
        int a = ((i - pos) << 1) + pos;
        int c = a + stride;
        bool asc = ((a & size) == 0);
        unsigned long long ka = key[a], kc = key[c];
        if ((ka > kc) == asc) { key[a] = kc; key[c] = ka; }
      }
      __syncthreads();
    }
  }
  for (int i = threadIdx.x; i < Nreal; i += 1024) {
    int s = (int)(key[i] & 0x3FFFull);
    sticker[(long)row * Nreal + i] = s;
    undo[(long)row * Nreal + s] = i;
  }
}

// ------------------------------------------------- gather sorted q/k(norm)/v
__global__ __launch_bounds__(64)
void gather_kernel(const float* __restrict__ qkb, const float* __restrict__ vb,
                   const int* __restrict__ sticker, float* __restrict__ sqk,
                   float* __restrict__ sk, float* __restrict__ sv, int Nreal, int T) {
  int j = blockIdx.x, row = blockIdx.y, d = threadIdx.x;
  long base = (long)row * Nreal + j;
  int st = sticker[base] % T;
  int b = row >> 3, hd = row & 7;
  long src = ((long)b * T + st) * DIMC + hd * DH + d;
  float q = qkb[src], vv = vb[src];
  __shared__ float red[64];
  red[d] = q * q;
  __syncthreads();
  for (int s = 32; s > 0; s >>= 1) {
    if (d < s) red[d] += red[d + s];
    __syncthreads();
  }
  float nrm = fmaxf(sqrtf(red[0]), 1e-12f);
  sqk[base * DH + d] = q;
  sk[base * DH + d] = q / nrm;
  sv[base * DH + d] = vv;
}

// ------------------------------------------------ chunked LSH attention core
// LDS layout (bytes), all 16B aligned:
#define QH_OFF   0        // Qh: 64 x stride 72 halves  = 9216
#define KH_OFF   9216     // Kh: 128 x 72               = 18432
#define VT_OFF   27648    // Vt: 64(d) x stride136 (k)  = 17408
#define DOT_OFF  45056    // dots: 64 x 128 f32         = 32768
#define PH_OFF   77824    // Ph: 64 x stride 136        = 17408
#define QT_OFF   95232    // 64 ints
#define KT_OFF   95488    // 128 ints
#define ATTN_LDS 96000

__global__ __launch_bounds__(256)
void lsh_attn_kernel(const float* __restrict__ sqk, const float* __restrict__ sk,
                     const float* __restrict__ sv, const int* __restrict__ sticker,
                     float* __restrict__ bo, float* __restrict__ lseout,
                     int Nreal, int T, int nc, int causal) {
  extern __shared__ char smem[];
  _Float16* Qh = (_Float16*)(smem + QH_OFF);
  _Float16* Kh = (_Float16*)(smem + KH_OFF);
  _Float16* Vt = (_Float16*)(smem + VT_OFF);
  float*  dots = (float*)(smem + DOT_OFF);
  _Float16* Ph = (_Float16*)(smem + PH_OFF);
  int* qt = (int*)(smem + QT_OFF);
  int* kt = (int*)(smem + KT_OFF);

  int c = blockIdx.x, row = blockIdx.y;
  int cp = (c + nc - 1) % nc;                   // attend one chunk back (rolled)
  long base = (long)row * Nreal;
  int j0 = c * 64, jp = cp * 64;
  int tid = threadIdx.x;

  // Q: 64 rows x 64 d, vectorized
  for (int i = tid; i < 64 * 16; i += 256) {
    int r = i >> 4, c4 = (i & 15) * 4;
    float4 f = *(const float4*)&sqk[(base + j0 + r) * DH + c4];
    v4h h; h[0] = (_Float16)f.x; h[1] = (_Float16)f.y; h[2] = (_Float16)f.z; h[3] = (_Float16)f.w;
    *(v4h*)(Qh + r * 72 + c4) = h;
  }
  // K rows (normalized) + V transposed
  for (int i = tid; i < 128 * 16; i += 256) {
    int r = i >> 4, c4 = (i & 15) * 4;
    long src = (r < 64) ? (base + j0 + r) : (base + jp + (r - 64));
    float4 fk = *(const float4*)&sk[src * DH + c4];
    v4h h; h[0] = (_Float16)fk.x; h[1] = (_Float16)fk.y; h[2] = (_Float16)fk.z; h[3] = (_Float16)fk.w;
    *(v4h*)(Kh + r * 72 + c4) = h;
    float4 fv = *(const float4*)&sv[src * DH + c4];
    float fvv[4] = {fv.x, fv.y, fv.z, fv.w};
#pragma unroll
    for (int j = 0; j < 4; ++j) Vt[(c4 + j) * 136 + r] = (_Float16)fvv[j];
  }
  if (tid < 64)  qt[tid] = sticker[base + j0 + tid] % T;
  if (tid < 128) kt[tid] = sticker[(tid < 64) ? (base + j0 + tid) : (base + jp + tid - 64)] % T;
  __syncthreads();

  int wave = tid >> 5, lane = tid & 31, nn = lane & 15, hi = lane >> 4;

  // dots = (Q @ K^T) * d^-0.5 ; 4x8 tiles of 16x16, 4 per wave
  {
    int qi = wave >> 1, g = wave & 1;
    v16h a0 = load_frag(Qh + qi * 16 * 72, 72, 0);    // hoisted A fragments
    v16h a1 = load_frag(Qh + qi * 16 * 72, 72, 32);
#pragma unroll
    for (int nt = 0; nt < 4; ++nt) {
      int ni = g * 4 + nt;
      v8f acc = {};
      acc = wmma16(a0, load_frag(Kh + ni * 16 * 72, 72, 0), acc);
      acc = wmma16(a1, load_frag(Kh + ni * 16 * 72, 72, 32), acc);
#pragma unroll
      for (int v = 0; v < 8; ++v)
        dots[(qi * 16 + v + 8 * hi) * 128 + ni * 16 + nn] = acc[v] * 0.125f;
    }
  }
  __syncthreads();

  // masked softmax + LSE: 4 lanes per query row, shuffle reductions
  {
    int q = tid >> 2, part = tid & 3;
    int qp = qt[q];
    int c0 = part * 32;
    float mx = -3.0e38f;
    for (int k = c0; k < c0 + 32; ++k) {
      float v = dots[q * 128 + k];
      int kp = kt[k];
      if (causal && qp < kp) v = -1e9f;
      if (qp == kp) v = -5e4f;
      dots[q * 128 + k] = v;
      mx = fmaxf(mx, v);
    }
    mx = fmaxf(mx, __shfl_xor(mx, 1, 32));
    mx = fmaxf(mx, __shfl_xor(mx, 2, 32));
    float s = 0.f;
    for (int k = c0; k < c0 + 32; ++k) s += __expf(dots[q * 128 + k] - mx);
    s += __shfl_xor(s, 1, 32);
    s += __shfl_xor(s, 2, 32);
    float l = mx + __logf(s);
    if (part == 0) lseout[base + j0 + q] = l;
#pragma unroll
    for (int k8 = 0; k8 < 32; k8 += 8) {
      v8h h;
#pragma unroll
      for (int j = 0; j < 8; ++j)
        h[j] = (_Float16)__expf(dots[q * 128 + c0 + k8 + j] - l);
      *(v8h*)(Ph + q * 136 + c0 + k8) = h;
    }
  }
  __syncthreads();

  // bo = P @ V ; 4x4 tiles of 16x16, 2 per wave
  {
    int mi = wave >> 1, g = wave & 1;
    v16h a0 = load_frag(Ph + mi * 16 * 136, 136, 0);   // hoisted P fragments
    v16h a1 = load_frag(Ph + mi * 16 * 136, 136, 32);
    v16h a2 = load_frag(Ph + mi * 16 * 136, 136, 64);
    v16h a3 = load_frag(Ph + mi * 16 * 136, 136, 96);
#pragma unroll
    for (int nt = 0; nt < 2; ++nt) {
      int ni = g * 2 + nt;
      v8f acc = {};
      acc = wmma16(a0, load_frag(Vt + ni * 16 * 136, 136, 0), acc);
      acc = wmma16(a1, load_frag(Vt + ni * 16 * 136, 136, 32), acc);
      acc = wmma16(a2, load_frag(Vt + ni * 16 * 136, 136, 64), acc);
      acc = wmma16(a3, load_frag(Vt + ni * 16 * 136, 136, 96), acc);
#pragma unroll
      for (int v = 0; v < 8; ++v)
        bo[(base + j0 + mi * 16 + v + 8 * hi) * DH + ni * 16 + nn] = acc[v];
    }
  }
}

// ----------------------------------------- unsort + combine two hash rounds
__global__ __launch_bounds__(64)
void unsort_kernel(const float* __restrict__ bo, const float* __restrict__ lse,
                   const int* __restrict__ undo, float* __restrict__ ocomb,
                   int Nreal, int T) {
  int t = blockIdx.x, row = blockIdx.y, d = threadIdx.x;
  long rb = (long)row * Nreal;
  int u0 = undo[rb + t], u1 = undo[rb + T + t];
  float l0 = lse[rb + u0], l1 = lse[rb + u1];
  float mx = fmaxf(l0, l1);
  float e0 = __expf(l0 - mx), e1 = __expf(l1 - mx);
  float inv = 1.f / (e0 + e1);
  float o = (e0 * bo[(rb + u0) * DH + d] + e1 * bo[(rb + u1) * DH + d]) * inv;
  int b = row >> 3, hd = row & 7;
  ocomb[((long)b * T + t) * DIMC + hd * DH + d] = o;
}

// ===========================================================================
extern "C" void kernel_launch(void* const* d_in, const int* in_sizes, int n_in,
                              void* d_out, int out_size, void* d_ws, size_t ws_size,
                              hipStream_t stream) {
  const float* x    = (const float*)d_in[0];
  const float* keys = (const float*)d_in[1];
  const float* ln_g = (const float*)d_in[2];
  const float* ln_b = (const float*)d_in[3];
  const float* Wqk  = (const float*)d_in[4];
  const float* Wv   = (const float*)d_in[5];
  const float* Wo   = (const float*)d_in[6];
  const float* boW  = (const float*)d_in[7];
  const float* W1   = (const float*)d_in[8];
  const float* b1   = (const float*)d_in[9];
  const float* W2   = (const float*)d_in[10];
  const float* b2   = (const float*)d_in[11];

  float* ws = (float*)d_ws;
  size_t off = 0;
  auto alloc = [&](size_t n) { float* p = ws + off; off += (n + 255) & ~(size_t)255; return p; };

  const long NX = (long)BQ * TSELF * DIMC;       // 4,194,304
  float* x1p   = alloc(NX);
  float* x2p   = alloc(NX);
  float* xx    = alloc((size_t)BQ * TKV * DIMC);
  float* qkb   = alloc((size_t)BQ * TKV * DIMC);
  float* vb    = alloc((size_t)BQ * TKV * DIMC);
  float* rot   = alloc(64 * 2 * 40);
  int* bucketsI = (int*)alloc((size_t)BH * 2 * TKV);
  int* stickerI = (int*)alloc((size_t)BH * 2 * TKV);
  int* undoI    = (int*)alloc((size_t)BH * 2 * TKV);
  float* sqk  = alloc((size_t)BH * 2 * TKV * DH);
  float* skn  = alloc((size_t)BH * 2 * TKV * DH);
  float* sv   = alloc((size_t)BH * 2 * TKV * DH);
  float* bo   = alloc((size_t)BH * 2 * TKV * DH);
  float* lse  = alloc((size_t)BH * 2 * TKV);
  float* ocomb = alloc((size_t)BQ * TKV * DIMC);
  float* proj  = alloc((size_t)BQ * TKV * DIMC);
  float* hidden = sqk;  // alias: FFN hidden (16.8M floats) reuses sqk+skn region

  auto gemm = [&](const float* A, const float* Bm, const float* bias, float* C,
                  const float* res, int M, int N, int K, int act) {
    dim3 g(N / 64, (M + 63) / 64);
    gemm_wmma_kernel<<<g, 256, 0, stream>>>(A, Bm, bias, C, res, M, N, K, act);
  };

  {
    int nb = (int)((NX + 255) / 256);
    copy_kernel<<<nb, 256, 0, stream>>>(x1p, x, NX);
    copy_kernel<<<nb, 256, 0, stream>>>(x2p, x, NX);
  }

  for (int l = 0; l < 2; ++l) {
    for (int which = 0; which < 2; ++which) {
      const int T = (which == 0) ? TSELF : TKV;
      const int nb = T / 64, nr = nb / 2;
      const int Nreal = 2 * T, nc = 2 * nb;
      const int Mrows = BQ * T;
      const int causal = (which == 0) ? 1 : 0;
      const int seed = 2 * l + which;

      ln_kernel<<<BQ * TSELF, 128, 0, stream>>>(
          x2p, ln_g + (l * 3 + which) * DIMC, ln_b + (l * 3 + which) * DIMC,
          xx, TSELF, T);
      if (which == 1) {
        long nk = (long)BQ * 1024 * DIMC;
        copy_keys_kernel<<<(int)((nk + 255) / 256), 256, 0, stream>>>(xx, keys);
      }

      const float* Wq  = Wqk + (size_t)(l * 2 + which) * DIMC * DIMC;
      const float* Wvv = Wv  + (size_t)(l * 2 + which) * DIMC * DIMC;
      const float* Woo = Wo  + (size_t)(l * 2 + which) * DIMC * DIMC;
      const float* bob = boW + (size_t)(l * 2 + which) * DIMC;

      gemm(xx, Wq, nullptr, qkb, nullptr, Mrows, DIMC, DIMC, 0);
      gemm(xx, Wvv, nullptr, vb, nullptr, Mrows, DIMC, DIMC, 0);

      int nrot = 64 * 2 * nr;
      rotgen_kernel<<<(nrot + 255) / 256, 256, 0, stream>>>(rot, nrot, seed);
      bucket_kernel<<<(BH * 2 * T + 255) / 256, 256, 0, stream>>>(qkb, rot, bucketsI, T, nr, nb);

      int npad = 1;
      while (npad < Nreal) npad <<= 1;
      sort_kernel<<<BH, 1024, (size_t)npad * 8, stream>>>(bucketsI, stickerI, undoI, Nreal, npad, T);

      gather_kernel<<<dim3(Nreal, BH), 64, 0, stream>>>(qkb, vb, stickerI, sqk, skn, sv, Nreal, T);
      lsh_attn_kernel<<<dim3(nc, BH), 256, ATTN_LDS, stream>>>(
          sqk, skn, sv, stickerI, bo, lse, Nreal, T, nc, causal);
      unsort_kernel<<<dim3(T, BH), 64, 0, stream>>>(bo, lse, undoI, ocomb, Nreal, T);

      gemm(ocomb, Woo, bob, proj, nullptr, Mrows, DIMC, DIMC, 0);
      add_trunc_kernel<<<(int)((NX + 255) / 256), 256, 0, stream>>>(x1p, proj, T);

      { float* tmp = x1p; x1p = x2p; x2p = tmp; }  // ReversibleSwap
    }

    // FFN
    ln_kernel<<<BQ * TSELF, 128, 0, stream>>>(
        x2p, ln_g + (l * 3 + 2) * DIMC, ln_b + (l * 3 + 2) * DIMC, xx, TSELF, TSELF);
    gemm(xx, W1 + (size_t)l * DIMC * 2048, b1 + (size_t)l * 2048, hidden, nullptr,
         BQ * TSELF, 2048, DIMC, 1);
    gemm(hidden, W2 + (size_t)l * 2048 * DIMC, b2 + (size_t)l * DIMC, x1p, x1p,
         BQ * TSELF, DIMC, 2048, 0);
    { float* tmp = x1p; x1p = x2p; x2p = tmp; }
  }

  final_add_kernel<<<(int)((NX + 255) / 256), 256, 0, stream>>>((float*)d_out, x1p, x2p, NX);
}
